// EdgeDecoder_15298673509108
// MI455X (gfx1250) — compile-verified, matrix operand order checked by reading, not run on hardware
//
#include <hip/hip_runtime.h>
#include <hip/hip_bf16.h>

// ---- vector types matching the CDNA5 WMMA builtin signatures ----
typedef __attribute__((ext_vector_type(16))) __bf16 v16bf;
typedef __attribute__((ext_vector_type(8)))  __bf16 v8bf;
typedef __attribute__((ext_vector_type(8)))  float  v8f;
typedef __attribute__((ext_vector_type(4)))  float  vf4;

#define D_EMB 128
#define H_DIM 128
#define K0    256           // 2*D
// LDS layout (bytes)
#define W0_BASE 0           // 64 KB : 8 k-tiles x 8 n-tiles x 1KB (B-swizzled bf16)
#define W1_BASE 65536       // 32 KB : 4 k-tiles x 8 n-tiles x 1KB
#define H1_BASE 98304       // 32 KB : 8 waves  x 4KB (16x128 bf16, row-major)
#define LDS_BYTES 131072

// Byte offset of W[k][n] inside a swizzled weight region (region = 8 n-tiles wide).
// B-operand layout for v_wmma_f32_16x16x32_bf16 (32x16 bf16):
//   lane l holds column N = l%16; K chunks [kb..kb+7] and [kb+16..kb+23], kb = (l/16)*8.
// Per tile (1KB): chunk0 region [0,512), chunk1 region [512,1024), lane stride 16B.
__device__ __forceinline__ int swz_off(int k, int n) {
  int kt = k >> 5, kk = k & 31;
  int nt = n >> 4, nn = n & 15;
  int hi    = (kk >> 3) & 1;
  int chunk = kk >> 4;
  int pos   = kk & 7;
  int lane  = hi * 16 + nn;
  return ((kt * 8 + nt) << 10) + (chunk << 9) + (lane << 4) + (pos << 1);
}

__device__ __forceinline__ v16bf load_B(const char* lds_byte, int base, int kt, int nt, int lane) {
  const char* tb = lds_byte + base + ((kt * 8 + nt) << 10) + (lane << 4);
  v8bf lo = *(const v8bf*)(tb);
  v8bf hi = *(const v8bf*)(tb + 512);
  v16bf B;
#pragma unroll
  for (int i = 0; i < 8; ++i) { B[i] = lo[i]; B[8 + i] = hi[i]; }
  return B;
}

__global__ void __launch_bounds__(256)
edge_mlp_wmma(const float* __restrict__ zu, const float* __restrict__ zm,
              const int* __restrict__ row, const int* __restrict__ col,
              const float* __restrict__ W0, const float* __restrict__ b0,
              const float* __restrict__ W1, const float* __restrict__ b1,
              const float* __restrict__ W2, const float* __restrict__ b2,
              float* __restrict__ out, int E, int nTiles) {
  extern __shared__ char lds[];

  const int tid = threadIdx.x;

  // ---- one-time: stage W0/W1 into LDS as bf16, pre-swizzled into B-operand layout ----
  for (int idx = tid; idx < K0 * H_DIM; idx += 256) {
    int k = idx >> 7, n = idx & 127;
    *(__bf16*)(lds + W0_BASE + swz_off(k, n)) = (__bf16)W0[idx];
  }
  for (int idx = tid; idx < H_DIM * H_DIM; idx += 256) {
    int k = idx >> 7, n = idx & 127;
    *(__bf16*)(lds + W1_BASE + swz_off(k, n)) = (__bf16)W1[idx];
  }
  __syncthreads();

  const int lane = tid & 31;
  const int wave = tid >> 5;
  const int nn   = lane & 15;   // A-row (edge within tile) this lane gathers / B column
  const int hi   = lane >> 4;   // K-half selector
  const int kb   = hi * 8;

  // per-lane invariant small vectors (hit L1/L2)
  float b0v[8], b1v[8], w2v[8];
#pragma unroll
  for (int nt = 0; nt < 8; ++nt) {
    b0v[nt] = b0[nt * 16 + nn];
    b1v[nt] = b1[nt * 16 + nn];
    w2v[nt] = W2[nt * 16 + nn];
  }
  const float b2s = b2[0];

  const v8f vzero = {0.f, 0.f, 0.f, 0.f, 0.f, 0.f, 0.f, 0.f};
  __bf16* h1 = (__bf16*)(lds + H1_BASE + wave * 4096);   // per-wave 16x128 bf16

  // ---- persistent loop over 16-edge tiles (128 edges per workgroup pass) ----
  for (int tile = blockIdx.x; tile < nTiles; tile += gridDim.x) {
    const long long eBase = (long long)tile * 128 + (long long)wave * 16;
    long long e  = eBase + nn;
    long long eg = (e < (long long)E) ? e : 0;     // clamp tail (EXEC must stay all-1s)
    const float* uPtr = zu + (long long)row[eg] * D_EMB + kb;
    const float* mPtr = zm + (long long)col[eg] * D_EMB + kb;

    // ================= Layer 0: [16,256] x [256,128] =================
    v8f acc[8];
#pragma unroll
    for (int nt = 0; nt < 8; ++nt) acc[nt] = vzero;

#pragma unroll
    for (int kt = 0; kt < 8; ++kt) {
      const float* src = (kt < 4) ? (uPtr + kt * 32) : (mPtr + (kt - 4) * 32);
      vf4 c0a = *(const vf4*)(src + 0);
      vf4 c0b = *(const vf4*)(src + 4);
      vf4 c1a = *(const vf4*)(src + 16);
      vf4 c1b = *(const vf4*)(src + 20);
      v16bf A;
#pragma unroll
      for (int i = 0; i < 4; ++i) {
        A[i]      = (__bf16)c0a[i];
        A[4 + i]  = (__bf16)c0b[i];
        A[8 + i]  = (__bf16)c1a[i];
        A[12 + i] = (__bf16)c1b[i];
      }
#pragma unroll
      for (int nt = 0; nt < 8; ++nt) {
        v16bf B = load_B(lds, W0_BASE, kt, nt, lane);
        acc[nt] = __builtin_amdgcn_wmma_f32_16x16x32_bf16(
            false, A, false, B, (short)0, acc[nt], false, false);
      }
    }

    // bias + ReLU -> bf16 h1 in LDS (D layout: lane gives N, vgpr gives M)
#pragma unroll
    for (int nt = 0; nt < 8; ++nt) {
#pragma unroll
      for (int v = 0; v < 8; ++v) {
        float x = fmaxf(acc[nt][v] + b0v[nt], 0.f);
        h1[(v + 8 * hi) * H_DIM + nt * 16 + nn] = (__bf16)x;
      }
    }
    asm volatile("s_wait_dscnt 0" ::: "memory");  // per-wave LDS RAW fence (in-order anyway)

    // ================= Layer 1: [16,128] x [128,128] =================
    v8f acc2[8];
#pragma unroll
    for (int nt = 0; nt < 8; ++nt) acc2[nt] = vzero;

    const __bf16* h1r = h1 + nn * H_DIM;  // this lane's A-row (M = nn)
#pragma unroll
    for (int kt = 0; kt < 4; ++kt) {
      v8bf alo = *(const v8bf*)(h1r + kt * 32 + kb);
      v8bf ahi = *(const v8bf*)(h1r + kt * 32 + kb + 16);
      v16bf A;
#pragma unroll
      for (int i = 0; i < 8; ++i) { A[i] = alo[i]; A[8 + i] = ahi[i]; }
#pragma unroll
      for (int nt = 0; nt < 8; ++nt) {
        v16bf B = load_B(lds, W1_BASE, kt, nt, lane);
        acc2[nt] = __builtin_amdgcn_wmma_f32_16x16x32_bf16(
            false, A, false, B, (short)0, acc2[nt], false, false);
      }
    }

    // ========== Layer 2: ReLU then dot with W2, 16-lane tree reduction ==========
    float s[8];
#pragma unroll
    for (int v = 0; v < 8; ++v) s[v] = 0.f;
#pragma unroll
    for (int nt = 0; nt < 8; ++nt) {
#pragma unroll
      for (int v = 0; v < 8; ++v) {
        float x = fmaxf(acc2[nt][v] + b1v[nt], 0.f);
        s[v] += x * w2v[nt];
      }
    }
#pragma unroll
    for (int v = 0; v < 8; ++v) {
      float t = s[v];
      t += __shfl_xor(t, 1, 32);
      t += __shfl_xor(t, 2, 32);
      t += __shfl_xor(t, 4, 32);
      t += __shfl_xor(t, 8, 32);
      s[v] = t;                       // every lane in its 16-lane half has the full sum
    }
    if (nn == 0) {                    // lanes 0 and 16 write M = v and M = v+8
#pragma unroll
      for (int v = 0; v < 8; ++v) {
        long long eo = eBase + v + 8 * hi;
        if (eo < (long long)E) out[eo] = s[v] + b2s;
      }
    }
  }
}

extern "C" void kernel_launch(void* const* d_in, const int* in_sizes, int n_in,
                              void* d_out, int out_size, void* d_ws, size_t ws_size,
                              hipStream_t stream) {
  const float* zu  = (const float*)d_in[0];
  const float* zm  = (const float*)d_in[1];
  const int*   row = (const int*)d_in[2];
  const int*   col = (const int*)d_in[3];
  const float* W0  = (const float*)d_in[4];
  const float* b0  = (const float*)d_in[5];
  const float* W1  = (const float*)d_in[6];
  const float* b1  = (const float*)d_in[7];
  const float* W2  = (const float*)d_in[8];
  const float* b2  = (const float*)d_in[9];
  float* out = (float*)d_out;

  const int E = in_sizes[2];                 // number of edges
  const int nTiles = (E + 127) / 128;        // 128 edges per workgroup pass
  int grid = nTiles < 2048 ? nTiles : 2048;  // persistent WGs amortize weight staging

  (void)hipFuncSetAttribute((const void*)edge_mlp_wmma,
                            hipFuncAttributeMaxDynamicSharedMemorySize, LDS_BYTES);

  edge_mlp_wmma<<<grid, 256, LDS_BYTES, stream>>>(
      zu, zm, row, col, W0, b0, W1, b1, W2, b2, out, E, nTiles);
}
